// VectorQuantizer_55456617725954
// MI455X (gfx1250) — matrix operand order repaired and checked by reading, not compiled, not run on hardware
//
#include <hip/hip_runtime.h>

typedef __bf16 bf16_t;
typedef __attribute__((ext_vector_type(16))) __bf16 v16bf;
typedef __attribute__((ext_vector_type(8)))  __bf16 v8bf;
typedef __attribute__((ext_vector_type(8)))  float  v8f;

#define NUM_EMB   8192
#define EMB_DIM   256
#define NVEC      16384   // 16*32*32

// d_out offsets (in floats), tuple return order:
// quantized [16,256,32,32], loss, perplexity, indices [16384,1], soft_codes [N,K]
#define Q_OFF     0
#define LOSS_OFF  4194304
#define PERP_OFF  4194305
#define IDX_OFF   4194306
#define SOFT_OFF  4210690

union V16U { v16bf v; v8bf h[2]; };

// ---------------------------------------------------------------------------
// Zero the histogram counts (d_ws is poisoned by the harness).
// ---------------------------------------------------------------------------
__global__ void vq_init(float* __restrict__ counts) {
    int t = blockIdx.x * 256 + threadIdx.x;
    if (t < NUM_EMB) counts[t] = 0.f;
}

// ---------------------------------------------------------------------------
// L2-normalize codebook rows -> bf16, plus wn2[k] = ||wn||^2 (f32, pre-bf16).
// One wave per row; lane handles 8 contiguous channels (coalesced both ways).
// ---------------------------------------------------------------------------
__global__ __launch_bounds__(256) void vq_norm_w(
    const float* __restrict__ emb, bf16_t* __restrict__ wnb,
    float* __restrict__ wn2g)
{
    const int lane = threadIdx.x & 31;
    const int wv   = threadIdx.x >> 5;
    const int row  = blockIdx.x * 8 + wv;
    const float* src = emb + (size_t)row * EMB_DIM + lane * 8;
    float v[8]; float ss = 0.f;
    #pragma unroll
    for (int j = 0; j < 8; ++j) { v[j] = src[j]; ss += v[j] * v[j]; }
    #pragma unroll
    for (int m = 1; m < 32; m <<= 1) ss += __shfl_xor(ss, m, 32);
    const float nrm = sqrtf(ss);
    const float iv  = 1.0f / fmaxf(nrm, 1e-12f);
    if (lane == 0) wn2g[row] = ss * iv * iv;
    bf16_t* dst = wnb + (size_t)row * EMB_DIM + lane * 8;
    #pragma unroll
    for (int j = 0; j < 8; ++j) dst[j] = (bf16_t)(v[j] * iv);
}

// ---------------------------------------------------------------------------
// Gather flat_x rows (NHWC view of NCHW input), L2-normalize -> bf16,
// plus fn2[n]. Block per (b,h); thread t: w = t&31 (its row), cg = t>>5
// (its 32-channel group). Input reads are w-coalesced (128B lines).
// ---------------------------------------------------------------------------
__global__ __launch_bounds__(256) void vq_norm_x(
    const float* __restrict__ in, bf16_t* __restrict__ fnb,
    float* __restrict__ fn2g)
{
    const int b  = blockIdx.x >> 5;
    const int h  = blockIdx.x & 31;
    const int t  = threadIdx.x;
    const int w  = t & 31;
    const int cg = t >> 5;
    const float* base = in + (size_t)b * (EMB_DIM * 1024) + (size_t)h * 32 + w;
    float x[32]; float ss = 0.f;
    #pragma unroll
    for (int j = 0; j < 32; ++j) {
        x[j] = base[(size_t)(cg * 32 + j) * 1024];
        ss  += x[j] * x[j];
    }
    __shared__ float part[8][32];
    __shared__ float sInv[32];
    part[cg][w] = ss;
    __syncthreads();
    if (t < 32) {
        float s = 0.f;
        #pragma unroll
        for (int g = 0; g < 8; ++g) s += part[g][t];
        const float nrm = sqrtf(s);
        const float iv  = 1.0f / fmaxf(nrm, 1e-12f);
        sInv[t] = iv;
        fn2g[b * 1024 + h * 32 + t] = s * iv * iv;
    }
    __syncthreads();
    const float iv = sInv[w];
    bf16_t* dst = fnb + (size_t)(b * 1024 + h * 32 + w) * EMB_DIM + cg * 32;
    #pragma unroll
    for (int j = 0; j < 32; ++j) dst[j] = (bf16_t)(x[j] * iv);
}

// ---------------------------------------------------------------------------
// Fused distance-GEMM + softmax + argmin.
// Block: 256 threads = 8 waves = 4 row-tiles x 2 K-halves. A-tile in regs.
// logit = -(fn2 + wn2 - 2*dot); bounded, so no max-subtraction needed.
// Pass 1: row exp-sums + argmax. LDS split-K merge. Pass 2: recompute GEMM,
// stream normalized probabilities with nontemporal stores.
// ---------------------------------------------------------------------------
__global__ __launch_bounds__(256) void vq_gemm_softmax(
    const bf16_t* __restrict__ fnb, const bf16_t* __restrict__ wnb,
    const float*  __restrict__ fn2g, const float* __restrict__ wn2g,
    float* __restrict__ soft_out, float* __restrict__ idxf_out,
    int* __restrict__ idxi_out, float* __restrict__ counts)
{
    const int lane = threadIdx.x & 31;
    const int wv   = threadIdx.x >> 5;   // 0..7
    const int rt   = wv & 3;             // row-tile in block
    const int kh   = wv >> 2;            // K half (0/1)
    const int half = lane >> 4;          // A/B/C lane-half
    const int r    = lane & 15;

    const int rowBase = blockIdx.x * 64 + rt * 16;

    // ---- A tile (16 rows x 256 ch bf16) resident in registers ----
    // 16-bit A 16x32 layout: lane half h, VGPRs 0-3 -> K = h*8 + 0..7,
    // VGPRs 4-7 -> K = 16 + h*8 + 0..7 : two contiguous 16B chunks.
    v16bf A[8];
    const bf16_t* arow = fnb + (size_t)(rowBase + r) * EMB_DIM;
    #pragma unroll
    for (int ks = 0; ks < 8; ++ks) {
        V16U u;
        u.h[0] = *reinterpret_cast<const v8bf*>(arow + ks * 32 + half * 8);
        u.h[1] = *reinterpret_cast<const v8bf*>(arow + ks * 32 + 16 + half * 8);
        A[ks] = u.v;
    }

    float f2[8];
    #pragma unroll
    for (int i = 0; i < 8; ++i) f2[i] = fn2g[rowBase + i + 8 * half];

    const int ctBase  = kh * 256;        // 256 col-tiles of 16 per K-half
    const int khalf16 = half * 8;        // B: lanes 16-31 hold K=8..15, 24..31

    float runSum[8], bestV[8]; int bestI[8];
    #pragma unroll
    for (int i = 0; i < 8; ++i) { runSum[i] = 0.f; bestV[i] = -3.402823466e38f; bestI[i] = 0; }

    // ---------------- pass 1: sums + argmax ----------------
    for (int ct = ctBase; ct < ctBase + 256; ++ct) {
        const int kcb = ct * 16 + r;                 // this lane's codebook col
        const bf16_t* bcol = wnb + (size_t)kcb * EMB_DIM + khalf16;
        v8f acc = {};
        #pragma unroll
        for (int ks = 0; ks < 8; ++ks) {
            V16U bb;
            bb.h[0] = *reinterpret_cast<const v8bf*>(bcol + ks * 32);
            bb.h[1] = *reinterpret_cast<const v8bf*>(bcol + ks * 32 + 16);
            acc = __builtin_amdgcn_wmma_f32_16x16x32_bf16(
                false, A[ks], false, bb.v, (short)0, acc, false, false);
        }
        const float w2 = wn2g[kcb];
        #pragma unroll
        for (int i = 0; i < 8; ++i) {
            const float v = 2.f * acc[i] - f2[i] - w2; // = -distance
            runSum[i] += __expf(v);
            if (v > bestV[i]) { bestV[i] = v; bestI[i] = kcb; } // first-min tiebreak
        }
    }

    // reduce across the 16 lanes of each half (rows i+8*half, cols = lanes)
    #pragma unroll
    for (int i = 0; i < 8; ++i) {
        float s = runSum[i], bv = bestV[i]; int bi = bestI[i];
        #pragma unroll
        for (int m = 1; m < 16; m <<= 1) {
            s += __shfl_xor(s, m, 32);
            float ov = __shfl_xor(bv, m, 32);
            int   oi = __shfl_xor(bi, m, 32);
            if (ov > bv || (ov == bv && oi < bi)) { bv = ov; bi = oi; }
        }
        runSum[i] = s; bestV[i] = bv; bestI[i] = bi;
    }

    __shared__ float sSum[2][64];
    __shared__ float sBV[2][64];
    __shared__ int   sBI[2][64];
    __shared__ float sInv[64];

    #pragma unroll
    for (int i = 0; i < 8; ++i) {
        if (r == i) {
            const int rb = rt * 16 + i + 8 * half;
            sSum[kh][rb] = runSum[i];
            sBV[kh][rb]  = bestV[i];
            sBI[kh][rb]  = bestI[i];
        }
    }
    __syncthreads();

    #pragma unroll
    for (int i = 0; i < 8; ++i) {
        if (kh == 0 && r == i) {
            const int rb = rt * 16 + i + 8 * half;
            const float tot = sSum[0][rb] + sSum[1][rb];
            float bv = sBV[0][rb]; int bi = sBI[0][rb];
            if (sBV[1][rb] > bv) { bv = sBV[1][rb]; bi = sBI[1][rb]; } // half0 idx < half1
            sInv[rb] = 1.0f / tot;
            const int rowg = rowBase + i + 8 * half;
            idxf_out[rowg] = (float)bi;
            idxi_out[rowg] = bi;
            atomicAdd(&counts[bi], 1.0f);   // integer-valued: order-independent
        }
    }
    __syncthreads();

    float inv[8];
    #pragma unroll
    for (int i = 0; i < 8; ++i) inv[i] = sInv[rt * 16 + i + 8 * half];

    // ---------------- pass 2: recompute + streamed probability writes ------
    for (int ct = ctBase; ct < ctBase + 256; ++ct) {
        const int kcb = ct * 16 + r;
        const bf16_t* bcol = wnb + (size_t)kcb * EMB_DIM + khalf16;
        v8f acc = {};
        #pragma unroll
        for (int ks = 0; ks < 8; ++ks) {
            V16U bb;
            bb.h[0] = *reinterpret_cast<const v8bf*>(bcol + ks * 32);
            bb.h[1] = *reinterpret_cast<const v8bf*>(bcol + ks * 32 + 16);
            acc = __builtin_amdgcn_wmma_f32_16x16x32_bf16(
                false, A[ks], false, bb.v, (short)0, acc, false, false);
        }
        const float w2 = wn2g[kcb];
        #pragma unroll
        for (int i = 0; i < 8; ++i) {
            const float p = __expf(2.f * acc[i] - f2[i] - w2) * inv[i];
            const size_t off = (size_t)(rowBase + i + 8 * half) * NUM_EMB + kcb;
            __builtin_nontemporal_store(p, &soft_out[off]);
        }
    }
}

// ---------------------------------------------------------------------------
// quantized[b,c,w,h] = emb[idx[b,h,w], c]; per-block partial of (q - x)^2
// written deterministically (no float atomics).
// ---------------------------------------------------------------------------
__global__ __launch_bounds__(256) void vq_quant_loss(
    const float* __restrict__ in, const float* __restrict__ emb,
    const int* __restrict__ idx, float* __restrict__ qout,
    float* __restrict__ blockSums)
{
    const int t = blockIdx.x * 256 + threadIdx.x;
    const int b = t >> 18;
    const int c = (t >> 10) & 255;
    const int w = (t >> 5) & 31;
    const int h = t & 31;
    const int n = (b << 10) + (h << 5) + w;
    const int k = idx[n];
    const float q = emb[(size_t)k * EMB_DIM + c];
    qout[t] = q;
    const float x = in[(b << 18) + (c << 10) + (h << 5) + w];
    float d = q - x; d = d * d;
    #pragma unroll
    for (int m = 1; m < 32; m <<= 1) d += __shfl_xor(d, m, 32);
    __shared__ float red[8];
    if ((threadIdx.x & 31) == 0) red[threadIdx.x >> 5] = d;
    __syncthreads();
    if (threadIdx.x == 0) {
        float s = 0.f;
        #pragma unroll
        for (int i = 0; i < 8; ++i) s += red[i];
        blockSums[blockIdx.x] = s;
    }
}

// ---------------------------------------------------------------------------
// Final scalar reductions: loss and perplexity. Single block, fixed-order.
// ---------------------------------------------------------------------------
__global__ __launch_bounds__(256) void vq_finalize(
    const float* __restrict__ counts, const float* __restrict__ blockSums,
    float* __restrict__ out_loss, float* __restrict__ out_perp)
{
    const int t = threadIdx.x;
    float ls = 0.f;
    for (int j = t; j < 16384; j += 256) ls += blockSums[j];
    float ent = 0.f;
    for (int j = t; j < NUM_EMB; j += 256) {
        const float p = counts[j] * (1.0f / 16384.0f);
        ent -= p * logf(p + 1e-10f);
    }
    __shared__ float rl[256];
    __shared__ float re[256];
    rl[t] = ls; re[t] = ent;
    __syncthreads();
    for (int s = 128; s > 0; s >>= 1) {
        if (t < s) { rl[t] += rl[t + s]; re[t] += re[t + s]; }
        __syncthreads();
    }
    if (t == 0) {
        *out_loss = 0.25f * rl[0] / 4194304.0f;  // COMMITMENT_COST * mean
        *out_perp = expf(re[0]);
    }
}

// ---------------------------------------------------------------------------
extern "C" void kernel_launch(void* const* d_in, const int* in_sizes, int n_in,
                              void* d_out, int out_size, void* d_ws, size_t ws_size,
                              hipStream_t stream)
{
    (void)in_sizes; (void)n_in; (void)out_size; (void)ws_size;
    const float* in  = (const float*)d_in[0];   // [16,256,32,32]
    const float* emb = (const float*)d_in[1];   // [8192,256]
    float* out = (float*)d_out;

    char* ws = (char*)d_ws;
    bf16_t* fnb    = (bf16_t*)(ws);               //  8 MB normalized x (bf16)
    bf16_t* wnb    = (bf16_t*)(ws + 8388608);     //  4 MB normalized codebook
    float*  fn2g   = (float*) (ws + 12582912);    // 64 KB ||fn||^2
    float*  wn2g   = (float*) (ws + 12648448);    // 32 KB ||wn||^2
    int*    idxi   = (int*)   (ws + 12681216);    // 64 KB argmin indices
    float*  counts = (float*) (ws + 12746752);    // 32 KB histogram
    float*  bsums  = (float*) (ws + 12779520);    // 64 KB loss partials

    vq_init        <<<32,            256, 0, stream>>>(counts);
    vq_norm_w      <<<NUM_EMB / 8,   256, 0, stream>>>(emb, wnb, wn2g);
    vq_norm_x      <<<16 * 32,       256, 0, stream>>>(in, fnb, fn2g);
    vq_gemm_softmax<<<NVEC / 64,     256, 0, stream>>>(fnb, wnb, fn2g, wn2g,
                                                       out + SOFT_OFF, out + IDX_OFF,
                                                       idxi, counts);
    vq_quant_loss  <<<4194304 / 256, 256, 0, stream>>>(in, emb, idxi, out + Q_OFF, bsums);
    vq_finalize    <<<1,             256, 0, stream>>>(counts, bsums,
                                                       out + LOSS_OFF, out + PERP_OFF);
}
